// ProbAttention_3289944948847
// MI455X (gfx1250) — compile-verified, matrix operand order checked by reading, not run on hardware
//
#include <hip/hip_runtime.h>
#include <hip/hip_bf16.h>

// Problem constants (match reference)
#define B_    4
#define H_    8
#define L_    2048
#define D_    64
#define U_    40     // samples per query
#define NTOP  40     // top-k queries per head
#define NPAD  48     // padded to 3 x 16-row WMMA tiles

typedef __attribute__((ext_vector_type(16))) _Float16 v16h;
typedef __attribute__((ext_vector_type(8)))  float    v8f;

__device__ __forceinline__ v8f wmma_f32_16x16x32_f16(v16h a, v16h b, v8f c) {
  // (neg_a, A, neg_b, B, c_mod, C, reuse_a, reuse_b)
  return __builtin_amdgcn_wmma_f32_16x16x32_f16(false, a, false, b, (short)0, c,
                                                false, false);
}

// ---------------------------------------------------------------------------
// Kernel 1: sparsity metric  M[b,h,l] = max_s(q.k_idx) - sum_s(q.k_idx)/L
// One thread per (b,h,l). q row cached in VGPRs (16 x float4).
// ---------------------------------------------------------------------------
__global__ void prob_metric_kernel(const float* __restrict__ Q,
                                   const float* __restrict__ K,
                                   const int*   __restrict__ ridx,
                                   float*       __restrict__ M) {
  int t = blockIdx.x * blockDim.x + threadIdx.x;
  if (t >= B_ * H_ * L_) return;
  int l  = t % L_;
  int bh = t / L_;
  int h  = bh % H_;
  int b  = bh / H_;

  const float4* q4 = reinterpret_cast<const float4*>(
      Q + (((size_t)b * L_ + l) * H_ + h) * D_);
  float4 qv[16];
#pragma unroll
  for (int i = 0; i < 16; ++i) qv[i] = q4[i];

  float mx = -1e30f, sm = 0.0f;
  for (int s = 0; s < U_; ++s) {
    int ki = ridx[l * U_ + s];
    const float4* k4 = reinterpret_cast<const float4*>(
        K + (((size_t)b * L_ + ki) * H_ + h) * D_);
    float d = 0.0f;
#pragma unroll
    for (int i = 0; i < 16; ++i) {
      float4 kv = k4[i];
      d += qv[i].x * kv.x + qv[i].y * kv.y + qv[i].z * kv.z + qv[i].w * kv.w;
    }
    mx = fmaxf(mx, d);
    sm += d;
  }
  M[t] = mx - sm * (1.0f / (float)L_);
}

// ---------------------------------------------------------------------------
// Kernel 2: per-(b,h) top-40 via 40 iterative argmax passes over 2048 values.
// One 256-thread block per head. Pads indices 40..47 with L-1.
// ---------------------------------------------------------------------------
__global__ void topk_kernel(const float* __restrict__ M, int* __restrict__ tidx) {
  __shared__ float sv[L_];
  __shared__ float rv[256];
  __shared__ int   ri[256];
  int bh  = blockIdx.x;
  int tid = threadIdx.x;

  for (int i = tid; i < L_; i += 256) sv[i] = M[(size_t)bh * L_ + i];
  __syncthreads();

  for (int u = 0; u < NTOP; ++u) {
    float bv = -1e30f; int bi = 0;
    for (int i = tid; i < L_; i += 256) {
      float v = sv[i];
      if (v > bv) { bv = v; bi = i; }
    }
    rv[tid] = bv; ri[tid] = bi;
    __syncthreads();
    for (int s = 128; s > 0; s >>= 1) {
      if (tid < s) {
        if (rv[tid + s] > rv[tid] ||
            (rv[tid + s] == rv[tid] && ri[tid + s] < ri[tid])) {
          rv[tid] = rv[tid + s]; ri[tid] = ri[tid + s];
        }
      }
      __syncthreads();
    }
    if (tid == 0) {
      tidx[bh * NPAD + u] = ri[0];
      sv[ri[0]] = -1e30f;
    }
    __syncthreads();
  }
  if (tid < NPAD - NTOP) tidx[bh * NPAD + NTOP + tid] = L_ - 1;
}

// ---------------------------------------------------------------------------
// Kernel 3: default output = cumsum(V) over sequence dim, written in (B,L,H,D).
// One block per (b,h), 64 threads (one per d). Coalesced 256B per step.
// ---------------------------------------------------------------------------
__global__ void cumsum_kernel(const float* __restrict__ V, float* __restrict__ out) {
  int bh = blockIdx.x;
  int b = bh / H_, h = bh % H_;
  int d = threadIdx.x;
  size_t base = ((size_t)b * L_ * H_ + h) * D_ + d;
  float s = 0.0f;
  for (int l = 0; l < L_; ++l) {
    s += V[base + (size_t)l * H_ * D_];
    out[base + (size_t)l * H_ * D_] = s;
  }
}

// ---------------------------------------------------------------------------
// Kernel 4: flash-style attention for the 48 (padded) selected queries.
// One block per (b,h), 96 threads = 3 waves; wave w owns Q rows 16w..16w+15.
// Streams keys/values in 32-wide chunks: 4 score WMMAs + online softmax +
// LDS relayout of P into A-fragment form + 4 accumulate WMMAs per wave/iter.
// Live ranges kept tight (scoped bk pair, split bv halves, fused softmax) so
// the Q fragments and accumulators stay resident — no scratch spills.
// Overwrites the cumsum rows for the NTOP selected queries.
// ---------------------------------------------------------------------------
__global__ void __launch_bounds__(96, 1)
prob_attn_wmma_kernel(const float* __restrict__ Q,
                      const float* __restrict__ K,
                      const float* __restrict__ V,
                      const int*   __restrict__ tidx,
                      float*       __restrict__ out) {
  __shared__ _Float16 pb[3][16][32];   // per-wave P staging (f16)

  const int lane = threadIdx.x & 31;
  const int w    = threadIdx.x >> 5;           // wave id 0..2
  const int bh   = blockIdx.x;
  const int b = bh / H_, h = bh % H_;
  const int n     = lane & 15;                 // column within 16-wide tile
  const int kbase = (lane < 16) ? 0 : 8;       // K-offset for A/B fragments
  const int rbase = (lane < 16) ? 0 : 8;       // row base for C fragments
  const float NEG = -1e30f;

  const int* ti = tidx + bh * NPAD;

  // ---- A fragments: Q_sel rows for this wave (f16), K(d) = 0..31 / 32..63 --
  const int uA = w * 16 + n;                   // A-layout row = lane%16
  const float* qp = Q + (((size_t)b * L_ + ti[uA]) * H_ + h) * D_;
  v16h a0, a1;
#pragma unroll
  for (int j = 0; j < 8; ++j) {
    a0[j]     = (_Float16)qp[kbase + j];
    a0[8 + j] = (_Float16)qp[16 + kbase + j];
    a1[j]     = (_Float16)qp[32 + kbase + j];
    a1[8 + j] = (_Float16)qp[48 + kbase + j];
  }

  // per-lane row state (C layout: this lane owns rows rbase+0..7 of its tile)
  int rowIdx[8];
#pragma unroll
  for (int r = 0; r < 8; ++r) rowIdx[r] = ti[w * 16 + rbase + r];
  float mrow[8], lrow[8];
  v8f acc[4];
#pragma unroll
  for (int r = 0; r < 8; ++r) { mrow[r] = NEG; lrow[r] = 0.0f; }
#pragma unroll
  for (int t = 0; t < 4; ++t)
#pragma unroll
    for (int e = 0; e < 8; ++e) acc[t][e] = 0.0f;

  v8f zero;
#pragma unroll
  for (int e = 0; e < 8; ++e) zero[e] = 0.0f;

  for (int j0 = 0; j0 < L_; j0 += 32) {
    if (j0 + 32 < L_)  // stream next key chunk (global_prefetch_b8)
      __builtin_prefetch(K + (((size_t)b * L_ + j0 + 32 + n) * H_ + h) * D_, 0, 1);

    // ---- scores: two 16x16 tiles, K(contraction)=64; bk scoped per tile --
    v8f s[2];
#pragma unroll
    for (int ct = 0; ct < 2; ++ct) {
      const float* kp = K + (((size_t)b * L_ + j0 + ct * 16 + n) * H_ + h) * D_;
      v16h bk0, bk1;
#pragma unroll
      for (int e = 0; e < 16; ++e) {
        int km = (e < 8) ? (kbase + e) : (16 + kbase + (e - 8));
        bk0[e] = (_Float16)kp[km];
        bk1[e] = (_Float16)kp[32 + km];
      }
      s[ct] = wmma_f32_16x16x32_f16(a0, bk0, zero);
      s[ct] = wmma_f32_16x16x32_f16(a1, bk1, s[ct]);
    }

    // ---- fused scale + causal mask + online softmax (per row) ------------
    float fac[8];
#pragma unroll
    for (int r = 0; r < 8; ++r) {
      float v0 = s[0][r] * 0.125f;
      float v1 = s[1][r] * 0.125f;
      if (j0 + n      > rowIdx[r]) v0 = NEG;
      if (j0 + 16 + n > rowIdx[r]) v1 = NEG;
      float tm = fmaxf(v0, v1);
#pragma unroll
      for (int msk = 1; msk < 16; msk <<= 1)
        tm = fmaxf(tm, __shfl_xor(tm, msk, 32));
      float mn = fmaxf(mrow[r], tm);
      fac[r]  = __expf(mrow[r] - mn);
      mrow[r] = mn;
      float p0 = __expf(v0 - mn);
      float p1 = __expf(v1 - mn);
      float rs = p0 + p1;
#pragma unroll
      for (int msk = 1; msk < 16; msk <<= 1)
        rs += __shfl_xor(rs, msk, 32);
      lrow[r] = lrow[r] * fac[r] + rs;
      // P straight to LDS (C layout -> A-fragment relayout)
      pb[w][rbase + r][n]      = (_Float16)p0;
      pb[w][rbase + r][16 + n] = (_Float16)p1;
    }
#pragma unroll
    for (int t = 0; t < 4; ++t)
#pragma unroll
      for (int r = 0; r < 8; ++r) acc[t][r] *= fac[r];

    __syncthreads();
    v16h pfrag;
#pragma unroll
    for (int e = 0; e < 16; ++e) {
      int km = (e < 8) ? (kbase + e) : (16 + kbase + (e - 8));
      pfrag[e] = pb[w][n][km];
    }
    __syncthreads();

    // ---- P @ V: 32 keys x 64 d, two fragment-pairs at a time -------------
#pragma unroll
    for (int half = 0; half < 2; ++half) {
      v16h bv0, bv1;
#pragma unroll
      for (int e = 0; e < 16; ++e) {
        int km = (e < 8) ? (kbase + e) : (16 + kbase + (e - 8));
        const float* vp = V + (((size_t)b * L_ + j0 + km) * H_ + h) * D_ +
                          half * 32;
        bv0[e] = (_Float16)vp[n];
        bv1[e] = (_Float16)vp[16 + n];
      }
      acc[half * 2]     = wmma_f32_16x16x32_f16(pfrag, bv0, acc[half * 2]);
      acc[half * 2 + 1] = wmma_f32_16x16x32_f16(pfrag, bv1, acc[half * 2 + 1]);
    }
  }

  // ---- normalize + scatter the NTOP rows over the cumsum baseline --------
#pragma unroll
  for (int r = 0; r < 8; ++r) {
    int u = w * 16 + rbase + r;
    if (u < NTOP) {
      float inv = 1.0f / lrow[r];
      float* op = out + (((size_t)b * L_ + rowIdx[r]) * H_ + h) * D_;
#pragma unroll
      for (int t = 0; t < 4; ++t) op[t * 16 + n] = acc[t][r] * inv;
    }
  }
}

// ---------------------------------------------------------------------------
extern "C" void kernel_launch(void* const* d_in, const int* in_sizes, int n_in,
                              void* d_out, int out_size, void* d_ws, size_t ws_size,
                              hipStream_t stream) {
  const float* Q    = (const float*)d_in[0];
  const float* K    = (const float*)d_in[1];
  const float* V    = (const float*)d_in[2];
  const int*   ridx = (const int*)d_in[4];   // d_in[3] = attn_mask (unused)
  float* out = (float*)d_out;

  // workspace: [ M : B*H*L floats ][ top_idx : B*H*NPAD ints ]
  float* Mbuf = (float*)d_ws;
  int*   tbuf = (int*)((char*)d_ws + (size_t)B_ * H_ * L_ * sizeof(float));

  prob_metric_kernel<<<(B_ * H_ * L_ + 255) / 256, 256, 0, stream>>>(Q, K, ridx, Mbuf);
  topk_kernel<<<B_ * H_, 256, 0, stream>>>(Mbuf, tbuf);
  cumsum_kernel<<<B_ * H_, D_, 0, stream>>>(V, out);
  prob_attn_wmma_kernel<<<B_ * H_, 96, 0, stream>>>(Q, K, V, tbuf, out);
}
